// MultiHeadAttention_48842368090198
// MI455X (gfx1250) — matrix operand order repaired and checked by reading, not compiled, not run on hardware
//
#include <hip/hip_runtime.h>
#include <stdint.h>

// ---------------------------------------------------------------------------
// MHA forward for MI455X (gfx1250, wave32, WMMA 16x16x32 f16 -> f32)
// B=4, T=2048, C=512, H=8, DK=64
// Flash-attention stage uses GLOBAL_LOAD_ASYNC_TO_LDS_B128 (ASYNCcnt) with
// double-buffered K/V tiles shared by the 4 waves of a block.
// ---------------------------------------------------------------------------

#define C_MODEL 512
#define T_SEQ   2048
#define BATCH   4
#define HEADS   8
#define DK      64
#define MROWS   (BATCH * T_SEQ)   // 8192

typedef _Float16 v8h  __attribute__((ext_vector_type(8)));
typedef _Float16 v16h __attribute__((ext_vector_type(16)));
typedef float    v8f  __attribute__((ext_vector_type(8)));

__device__ __forceinline__ v8f wmma16(v16h a, v16h b, v8f c) {
  // D = A(16x32 f16) * B(32x16 f16) + C(16x16 f32)
  return __builtin_amdgcn_wmma_f32_16x16x32_f16(false, a, false, b, (short)0, c,
                                                false, false);
}

// A-fragment (16x32 f16), source row-major with leading dim `ld` (elements).
// ISA: lanes 0-15 hold K 0..7 & 16..23; lanes 16-31 hold K 8..15 & 24..31;
// row = lane & 15.
__device__ __forceinline__ v16h load_a(const _Float16* __restrict__ base,
                                       int ld, int lane) {
  const int m  = lane & 15;
  const int kb = (lane >> 4) << 3;
  const _Float16* p = base + m * ld;
  v8h lo = *(const v8h*)(p + kb);
  v8h hi = *(const v8h*)(p + 16 + kb);
  return __builtin_shufflevector(lo, hi, 0, 1, 2, 3, 4, 5, 6, 7, 8, 9, 10, 11,
                                 12, 13, 14, 15);
}

// B-fragment (32x16 f16), source stored column-contiguous over K: element
// [k][n] at base[n*ld + k]. ISA: lanes 0-15 hold K 0..7 & 16..23, lanes 16-31
// hold K 8..15 & 24..31; column = lane & 15. Two 16-byte slices per lane.
__device__ __forceinline__ v16h load_b(const _Float16* __restrict__ base,
                                       int ld, int lane) {
  const int n  = lane & 15;
  const int ks = (lane >> 4) << 3;
  const _Float16* p = base + n * ld + ks;
  v8h lo = *(const v8h*)(p);
  v8h hi = *(const v8h*)(p + 16);
  return __builtin_shufflevector(lo, hi, 0, 1, 2, 3, 4, 5, 6, 7, 8, 9, 10, 11,
                                 12, 13, 14, 15);
}

// One async 16B global->LDS copy (GV mode, ASYNCcnt-tracked). LDS operand is
// the low 32 bits of the generic pointer (= LDS byte offset per aperture).
__device__ __forceinline__ void async_cp16(const _Float16* gsrc, void* ldst) {
  const unsigned l = (unsigned)(uintptr_t)ldst;
  asm volatile("global_load_async_to_lds_b128 %0, %1, off" ::"v"(l), "v"(gsrc)
               : "memory");
}

// ---------------------------------------------------------------------------
// Stage 0: f32 -> f16 conversion (x and the four weight matrices)
// ---------------------------------------------------------------------------
__global__ void mha_cvt_f16_kernel(const float* __restrict__ src,
                                   _Float16* __restrict__ dst, int n) {
  int i = blockIdx.x * blockDim.x + threadIdx.x;
  const int stride = gridDim.x * blockDim.x;
  for (; i < n; i += stride) dst[i] = (_Float16)src[i];
}

// ---------------------------------------------------------------------------
// Stage 1: fused QKV projection. y = x @ w^T + b  (w is [out,in] row-major,
// so w rows ARE the B-matrix columns -> contiguous b128 fragment loads).
// Q,K written head-major [B,H,T,DK] (f16); V written transposed [B,H,DK,T].
// Wave owns a 16-row x 64-col tile. 3 mats * 512 row-tiles * 8 col-groups
// = 12288 waves = 3072 blocks of 128 threads.
// ---------------------------------------------------------------------------
__global__ __launch_bounds__(128) void mha_qkv_kernel(
    const _Float16* __restrict__ xh, const _Float16* __restrict__ wqh,
    const _Float16* __restrict__ wkh, const _Float16* __restrict__ wvh,
    const float* __restrict__ bq, const float* __restrict__ bk,
    const float* __restrict__ bv, _Float16* __restrict__ Qh,
    _Float16* __restrict__ Kh, _Float16* __restrict__ Vt) {
  const int lane = threadIdx.x & 31;
  const int wid  = (blockIdx.x << 2) | (threadIdx.x >> 5);
  const int mat  = wid >> 12;      // 0:Q 1:K 2:V
  const int rem  = wid & 4095;
  const int rt   = rem >> 3;       // row tile (16 rows), 0..511
  const int ctg  = rem & 7;        // 64-column group

  const _Float16* w = (mat == 0) ? wqh : (mat == 1) ? wkh : wvh;
  const float* bias = (mat == 0) ? bq : (mat == 1) ? bk : bv;

  const _Float16* arow = xh + (size_t)rt * 16 * C_MODEL;
  const int n0 = ctg * 64;

  v8f c0 = {}, c1 = {}, c2 = {}, c3 = {};
  for (int k0 = 0; k0 < C_MODEL; k0 += 32) {
    __builtin_prefetch(arow + (lane & 15) * C_MODEL + k0 + 32, 0, 1);
    v16h a  = load_a(arow + k0, C_MODEL, lane);
    v16h b0 = load_b(w + (size_t)(n0 + 0) * C_MODEL + k0, C_MODEL, lane);
    v16h b1 = load_b(w + (size_t)(n0 + 16) * C_MODEL + k0, C_MODEL, lane);
    v16h b2 = load_b(w + (size_t)(n0 + 32) * C_MODEL + k0, C_MODEL, lane);
    v16h b3 = load_b(w + (size_t)(n0 + 48) * C_MODEL + k0, C_MODEL, lane);
    c0 = wmma16(a, b0, c0);
    c1 = wmma16(a, b1, c1);
    c2 = wmma16(a, b2, c2);
    c3 = wmma16(a, b3, c3);
  }

  const int half  = lane >> 4;
  const int nl    = lane & 15;
  const int trow0 = rt * 16 + half * 8;  // global row base for this lane half
  const int bIdx  = trow0 >> 11;         // batch (T=2048 rows per batch)
  const int tt0   = trow0 & 2047;        // token within batch

  v8f cc[4] = {c0, c1, c2, c3};
#pragma unroll
  for (int ct = 0; ct < 4; ++ct) {
    const int n    = n0 + ct * 16 + nl;  // output feature 0..511
    const float bb = bias[n];
    const int h = n >> 6, d = n & 63;
    if (mat == 2) {
      // Vt[b][h][d][t]: 8 consecutive tokens -> one b128 store
      _Float16* dst = Vt + (((size_t)(bIdx * HEADS + h) * DK + d) << 11) + tt0;
      v8h ov;
#pragma unroll
      for (int r = 0; r < 8; ++r) ov[r] = (_Float16)(cc[ct][r] + bb);
      *(v8h*)dst = ov;
    } else {
      _Float16* mOut = (mat == 0) ? Qh : Kh;
      _Float16* dst =
          mOut + (((size_t)(bIdx * HEADS + h) * T_SEQ + tt0) << 6) + d;
#pragma unroll
      for (int r = 0; r < 8; ++r)
        dst[(size_t)r << 6] = (_Float16)(cc[ct][r] + bb);
    }
  }
}

// ---------------------------------------------------------------------------
// Stage 2: flash attention. Block = 4 waves = 4 adjacent 16-query tiles of
// one (b,h). Per 32-key step the block cooperatively async-copies the K tile
// (32x64) and V tile (64x32, from transposed V) into double-buffered LDS
// (4 GLOBAL_LOAD_ASYNC_TO_LDS_B128 per thread), overlapping the next tile's
// DMA with this tile's WMMAs; s_wait_asynccnt + barrier ping-pong. Softmax is
// online (half-wave shfl reductions match the C-fragment row layout); the P
// tile converts C-layout f32 -> A-layout f16 through per-wave LDS.
// Grid: 32 (b,h) * 32 query-blocks = 1024 blocks of 128.
// ---------------------------------------------------------------------------
__global__ __launch_bounds__(128) void mha_flash_attn_kernel(
    const _Float16* __restrict__ Qh, const _Float16* __restrict__ Kh,
    const _Float16* __restrict__ Vt, _Float16* __restrict__ Ah) {
  // Padded strides: 72 -> 36*n mod 64 bank spread, 40 -> 20*n mod 64 spread;
  // both keep every 16B fragment chunk 16B-aligned.
  __shared__ __align__(16) _Float16 ldsK[2][32][72];
  __shared__ __align__(16) _Float16 ldsV[2][64][40];
  __shared__ __align__(16) _Float16 ldsP[4][16][40];

  const int wslot = threadIdx.x >> 5;
  const int lane  = threadIdx.x & 31;
  const int bh    = blockIdx.x >> 5;          // b*8+h
  const int qblk  = blockIdx.x & 31;
  const int qt    = (qblk << 2) | wslot;      // this wave's 16-query tile

  const _Float16* Qbase = Qh + (size_t)bh * T_SEQ * DK + (size_t)(qt * 16) * DK;
  const _Float16* Kbase = Kh + (size_t)bh * T_SEQ * DK;
  const _Float16* Vbase = Vt + (size_t)bh * DK * T_SEQ;

  // Cooperative staging slices (per thread: 2 K chunks + 2 V chunks of 16B)
  const int tid = threadIdx.x;
  const int kt0 = tid >> 3;              // K tile row (token) 0..15
  const int ko  = (tid & 7) << 3;        // f16 offset within 64-wide row
  const int vd0 = tid >> 2;              // V tile row (d) 0..31
  const int vo  = (tid & 3) << 3;        // f16 offset within 32-wide row

  auto stage = [&](int buf, int j0) {
    const _Float16* g0 = Kbase + (size_t)(j0 + kt0) * DK + ko;
    const _Float16* g1 = Kbase + (size_t)(j0 + kt0 + 16) * DK + ko;
    const _Float16* g2 = Vbase + (size_t)vd0 * T_SEQ + j0 + vo;
    const _Float16* g3 = Vbase + (size_t)(vd0 + 32) * T_SEQ + j0 + vo;
    async_cp16(g0, &ldsK[buf][kt0][ko]);
    async_cp16(g1, &ldsK[buf][kt0 + 16][ko]);
    async_cp16(g2, &ldsV[buf][vd0][vo]);
    async_cp16(g3, &ldsV[buf][vd0 + 32][vo]);
  };

  const v16h qa0 = load_a(Qbase + 0, DK, lane);   // d = 0..31
  const v16h qa1 = load_a(Qbase + 32, DK, lane);  // d = 32..63

  v8f o0 = {}, o1 = {}, o2 = {}, o3 = {};
  float mrow[8], lrow[8];
#pragma unroll
  for (int r = 0; r < 8; ++r) {
    mrow[r] = -1e30f;
    lrow[r] = 0.0f;
  }

  const float scale = 0.125f;  // 1/sqrt(64)
  const int pm  = lane & 15;
  const int pkb = (lane >> 4) << 3;

  stage(0, 0);  // prologue: tile 0 in flight

  for (int it = 0; it < T_SEQ / 32; ++it) {
    const int j0  = it * 32;
    const int buf = it & 1;
    if (it + 1 < T_SEQ / 32) {
      stage(buf ^ 1, j0 + 32);  // overlap next tile's DMA with compute
      asm volatile("s_wait_asynccnt 0x4" ::: "memory");
    } else {
      asm volatile("s_wait_asynccnt 0x0" ::: "memory");
    }
    __syncthreads();  // every wave's slice of tile `it` is in LDS

    // ---- scores: S = Q K^T for keys [j0, j0+32), fragments from LDS -----
    v16h kb0a = load_b(&ldsK[buf][0][0], 72, lane);
    v16h kb0b = load_b(&ldsK[buf][0][32], 72, lane);
    v16h kb1a = load_b(&ldsK[buf][16][0], 72, lane);
    v16h kb1b = load_b(&ldsK[buf][16][32], 72, lane);
    v8f s0 = {}, s1 = {};
    s0 = wmma16(qa0, kb0a, s0);
    s0 = wmma16(qa1, kb0b, s0);
    s1 = wmma16(qa0, kb1a, s1);
    s1 = wmma16(qa1, kb1b, s1);

    // ---- online softmax per row group (row = r + 8*(lane>>4)) -----------
#pragma unroll
    for (int r = 0; r < 8; ++r) {
      const float sa = s0[r] * scale;
      const float sb = s1[r] * scale;
      float mx = fmaxf(sa, sb);
      mx = fmaxf(mx, __shfl_xor(mx, 1, 32));
      mx = fmaxf(mx, __shfl_xor(mx, 2, 32));
      mx = fmaxf(mx, __shfl_xor(mx, 4, 32));
      mx = fmaxf(mx, __shfl_xor(mx, 8, 32));
      const float mnew  = fmaxf(mrow[r], mx);
      const float alpha = __expf(mrow[r] - mnew);
      const float p0    = __expf(sa - mnew);
      const float p1    = __expf(sb - mnew);
      float ps = p0 + p1;
      ps += __shfl_xor(ps, 1, 32);
      ps += __shfl_xor(ps, 2, 32);
      ps += __shfl_xor(ps, 4, 32);
      ps += __shfl_xor(ps, 8, 32);
      lrow[r] = lrow[r] * alpha + ps;
      mrow[r] = mnew;
      o0[r] *= alpha;
      o1[r] *= alpha;
      o2[r] *= alpha;
      o3[r] *= alpha;
      // stage P tile (16 queries x 32 keys) in per-wave LDS, row-major
      const int m = r + ((lane >> 4) << 3);
      const int n = lane & 15;
      ldsP[wslot][m][n]      = (_Float16)p0;
      ldsP[wslot][m][16 + n] = (_Float16)p1;
    }

    // ---- reload P as an A-fragment (ds_load_b128) -----------------------
    v8h plo = *(const v8h*)&ldsP[wslot][pm][pkb];
    v8h phi = *(const v8h*)&ldsP[wslot][pm][16 + pkb];
    v16h pa = __builtin_shufflevector(plo, phi, 0, 1, 2, 3, 4, 5, 6, 7, 8, 9,
                                      10, 11, 12, 13, 14, 15);

    // ---- O += P @ V, fragments from LDS (keys contiguous per d row) -----
    v16h vb0 = load_b(&ldsV[buf][0][0], 40, lane);
    v16h vb1 = load_b(&ldsV[buf][16][0], 40, lane);
    v16h vb2 = load_b(&ldsV[buf][32][0], 40, lane);
    v16h vb3 = load_b(&ldsV[buf][48][0], 40, lane);
    o0 = wmma16(pa, vb0, o0);
    o1 = wmma16(pa, vb1, o1);
    o2 = wmma16(pa, vb2, o2);
    o3 = wmma16(pa, vb3, o3);

    __syncthreads();  // all waves done reading buf before it is refilled
  }

  // ---- normalize and write Ah [B,T,C] (f16) -----------------------------
  const int b = bh >> 3, h = bh & 7;
  const size_t rowbase =
      (size_t)(b * T_SEQ + qt * 16 + ((lane >> 4) << 3)) * C_MODEL + h * DK +
      (lane & 15);
#pragma unroll
  for (int r = 0; r < 8; ++r) {
    const float inv = 1.0f / lrow[r];
    Ah[rowbase + (size_t)r * C_MODEL + 0]  = (_Float16)(o0[r] * inv);
    Ah[rowbase + (size_t)r * C_MODEL + 16] = (_Float16)(o1[r] * inv);
    Ah[rowbase + (size_t)r * C_MODEL + 32] = (_Float16)(o2[r] * inv);
    Ah[rowbase + (size_t)r * C_MODEL + 48] = (_Float16)(o3[r] * inv);
  }
}

// ---------------------------------------------------------------------------
// Stage 3: output projection: out = Ah @ wo^T + bo (f32 out).
// 512 row-tiles * 8 col-groups = 4096 waves = 1024 blocks.
// ---------------------------------------------------------------------------
__global__ __launch_bounds__(128) void mha_outproj_kernel(
    const _Float16* __restrict__ Ah, const _Float16* __restrict__ woh,
    const float* __restrict__ bo, float* __restrict__ out) {
  const int lane = threadIdx.x & 31;
  const int wid  = (blockIdx.x << 2) | (threadIdx.x >> 5);
  const int rt   = wid >> 3;
  const int ctg  = wid & 7;

  const _Float16* arow = Ah + (size_t)rt * 16 * C_MODEL;
  const int n0 = ctg * 64;

  v8f c0 = {}, c1 = {}, c2 = {}, c3 = {};
  for (int k0 = 0; k0 < C_MODEL; k0 += 32) {
    __builtin_prefetch(arow + (lane & 15) * C_MODEL + k0 + 32, 0, 1);
    v16h a  = load_a(arow + k0, C_MODEL, lane);
    v16h b0 = load_b(woh + (size_t)(n0 + 0) * C_MODEL + k0, C_MODEL, lane);
    v16h b1 = load_b(woh + (size_t)(n0 + 16) * C_MODEL + k0, C_MODEL, lane);
    v16h b2 = load_b(woh + (size_t)(n0 + 32) * C_MODEL + k0, C_MODEL, lane);
    v16h b3 = load_b(woh + (size_t)(n0 + 48) * C_MODEL + k0, C_MODEL, lane);
    c0 = wmma16(a, b0, c0);
    c1 = wmma16(a, b1, c1);
    c2 = wmma16(a, b2, c2);
    c3 = wmma16(a, b3, c3);
  }

  const int half = lane >> 4;
  const int nl   = lane & 15;
  v8f cc[4] = {c0, c1, c2, c3};
#pragma unroll
  for (int ct = 0; ct < 4; ++ct) {
    const int n    = n0 + ct * 16 + nl;
    const float bb = bo[n];
    float* dst = out + (size_t)(rt * 16 + half * 8) * C_MODEL + n;
#pragma unroll
    for (int r = 0; r < 8; ++r) dst[(size_t)r * C_MODEL] = cc[ct][r] + bb;
  }
}

// ---------------------------------------------------------------------------
// Host-side launcher
// ---------------------------------------------------------------------------
extern "C" void kernel_launch(void* const* d_in, const int* in_sizes, int n_in,
                              void* d_out, int out_size, void* d_ws,
                              size_t ws_size, hipStream_t stream) {
  const float* x  = (const float*)d_in[0];
  const float* wq = (const float*)d_in[1];
  const float* bq = (const float*)d_in[2];
  const float* wk = (const float*)d_in[3];
  const float* bk = (const float*)d_in[4];
  const float* wv = (const float*)d_in[5];
  const float* bv = (const float*)d_in[6];
  const float* wo = (const float*)d_in[7];
  const float* bo = (const float*)d_in[8];
  float* out = (float*)d_out;

  // Workspace carve-up (f16 elements)
  _Float16* ws  = (_Float16*)d_ws;
  const size_t NX = (size_t)MROWS * C_MODEL;       // 4,194,304
  const size_t NW = (size_t)C_MODEL * C_MODEL;     //   262,144
  _Float16* xh  = ws;
  _Float16* wqh = xh + NX;
  _Float16* wkh = wqh + NW;
  _Float16* wvh = wkh + NW;
  _Float16* woh = wvh + NW;
  _Float16* Qh  = woh + NW;
  _Float16* Kh  = Qh + NX;
  _Float16* Vt  = Kh + NX;
  _Float16* Ah  = Vt + NX;

  // Stage 0: conversions
  mha_cvt_f16_kernel<<<4096, 256, 0, stream>>>(x, xh, (int)NX);
  mha_cvt_f16_kernel<<<512, 256, 0, stream>>>(wq, wqh, (int)NW);
  mha_cvt_f16_kernel<<<512, 256, 0, stream>>>(wk, wkh, (int)NW);
  mha_cvt_f16_kernel<<<512, 256, 0, stream>>>(wv, wvh, (int)NW);
  mha_cvt_f16_kernel<<<512, 256, 0, stream>>>(wo, woh, (int)NW);

  // Stage 1: QKV projection (12288 waves / 4 per block)
  mha_qkv_kernel<<<3072, 128, 0, stream>>>(xh, wqh, wkh, wvh, bq, bk, bv, Qh,
                                           Kh, Vt);

  // Stage 2: flash attention (1024 blocks of 4 cooperating waves)
  mha_flash_attn_kernel<<<1024, 128, 0, stream>>>(Qh, Kh, Vt, Ah);

  // Stage 3: output projection (4096 waves / 4 per block)
  mha_outproj_kernel<<<1024, 128, 0, stream>>>(Ah, woh, bo, out);

  (void)in_sizes;
  (void)n_in;
  (void)out_size;
  (void)ws_size;
}
